// EstimateCentroids_26989574488093
// MI455X (gfx1250) — matrix operand order repaired and checked by reading, not compiled, not run on hardware
//
#include <hip/hip_runtime.h>
#include <stdint.h>

// ---------------- problem constants ----------------
#define D0 512
#define D1 512
#define D2 48
#define VOX (D0 * D1 * D2)          // 12,582,912 voxels
#define NROWS (D0 * D1)             // 262,144 rows of 48 along w
#define RAD 4                       // 4x (3^3 erosion) == 9-wide box min per axis
#define MCLUST 4096
#define CHUNK 4096                  // voxels per TDM tile
#define NCHUNK (VOX / CHUNK)        // 3072 (exact)
#define SCALE_F 25.0f
#define QOFF 128.0f
#define MINS 10.0f
#define BOXH 22.5f
#define IOUTHR 0.5f

typedef unsigned int u32x4 __attribute__((ext_vector_type(4)));
typedef int          i32x8 __attribute__((ext_vector_type(8)));
typedef int          i32x4 __attribute__((ext_vector_type(4)));
typedef _Float16     v16h  __attribute__((ext_vector_type(16)));
typedef float        v8f   __attribute__((ext_vector_type(8)));

#if defined(__has_builtin)
#if __has_builtin(__builtin_amdgcn_tensor_load_to_lds) && \
    __has_builtin(__builtin_amdgcn_s_wait_tensorcnt)
#define HAVE_TDM 1
#endif
#endif

#define ZERO16 {0,0,0,0,0,0,0,0,0,0,0,0,0,0,0,0}
#define ZERO8  {0,0,0,0,0,0,0,0}

// ---------------- w-axis erosion via WMMA band-matrix multiply ----------------
// out[row][w] = (sum_{|d|<=4, 0<=w+d<48} bin(pm[row][w+d]) == 9)
// Sliding-window sum == matmul with banded 0/1 matrix Band[64][48]
// (K zero-padded 48..63). Per 16x48 tile: Out = In(16x64) x Band, done as
// 2 chained v_wmma_f32_16x16x32_f16 per 16-wide output block (6 WMMAs/tile).
// Sums are exact small integers in f32; edge columns see <9 taps -> fail ==9,
// which is exactly the zero-border erosion semantics.
__global__ void __launch_bounds__(256)
erode_w_wmma_kernel(const float* __restrict__ pm, unsigned char* __restrict__ out) {
  __shared__ _Float16 tileLds[8][16][64];   // one 16x64 f16 tile per wave (16KB)

  const int lane = threadIdx.x & 31;
  const int wave = threadIdx.x >> 5;
  _Float16 (*T)[64] = tileLds[wave];

  // --- constant band-matrix operands, built once (B in contiguous k order, same as A:
  // lanes 0-15: rows K=2v,2v+1 at N=lane; lanes 16-31: rows K=16+2v.. at N=lane-16)
  const int nloc = lane & 15;
  const int kgrp = lane >> 4;
  v16h bmat[3][2];
#pragma unroll
  for (int n = 0; n < 3; ++n) {
#pragma unroll
    for (int h = 0; h < 2; ++h) {
      v16h b = ZERO16;
#pragma unroll
      for (int v = 0; v < 8; ++v) {
#pragma unroll
        for (int p = 0; p < 2; ++p) {
          int K = 32 * h + 16 * kgrp + 2 * v + p;   // input column w'
          int wout = 16 * n + nloc;                 // output column w
          int diff = K - wout;
          if (diff < 0) diff = -diff;
          b[2 * v + p] = (K < 48 && diff <= RAD) ? (_Float16)1.0f : (_Float16)0.0f;
        }
      }
      bmat[n][h] = b;
    }
  }

  const int m  = lane & 15;          // A row (per ISA 7.12.2 A-layout)
  const int kb = (lane >> 4) * 16;   // lanes 0-15: K 0..15; 16-31: 16..31

  const int tileStride = gridDim.x * 8;                 // wave-tiles per sweep
  for (int tile = blockIdx.x * 8 + wave; tile < NROWS / 16; tile += tileStride) {
    const int rowBase = tile * 16;

    // stage 16x48 floats, binarized to f16 {0,1}, K 48..63 zero-padded
    for (int idx = lane; idx < 16 * 64; idx += 32) {
      int r = idx >> 6, c = idx & 63;
      float v = (c < 48) ? pm[(rowBase + r) * 48 + c] : 0.0f;
      T[r][c] = (v > 0.5f) ? (_Float16)1.0f : (_Float16)0.0f;
    }
    __syncthreads();

    // gather A operands (K 0..31 and K 32..63)
    v16h alo = ZERO16, ahi = ZERO16;
#pragma unroll
    for (int j = 0; j < 8; ++j) {
      alo[j]     = T[m][kb + j];
      alo[j + 8] = T[m][kb + 8 + j];
      ahi[j]     = T[m][32 + kb + j];
      ahi[j + 8] = T[m][32 + kb + 8 + j];
    }

#pragma unroll
    for (int n = 0; n < 3; ++n) {
      v8f c = ZERO8;
      c = __builtin_amdgcn_wmma_f32_16x16x32_f16(false, alo, false, bmat[n][0],
                                                 (short)0, c, false, false);
      c = __builtin_amdgcn_wmma_f32_16x16x32_f16(false, ahi, false, bmat[n][1],
                                                 (short)0, c, false, false);
      // C/D layout: N = lane&15, M = r + 8*(lane>=16)
#pragma unroll
      for (int r = 0; r < 8; ++r) {
        int M = r + 8 * kgrp;
        out[(rowBase + M) * 48 + 16 * n + nloc] = (c[r] == 9.0f) ? 1 : 0;
      }
    }
    __syncthreads();   // WAR on the LDS tile before next iteration
  }
}

// ---------------- x/y-axis erosion (9-wide binary min, byte passes) ----------
__global__ void erode_axis_kernel(const unsigned char* __restrict__ in,
                                  unsigned char* __restrict__ out,
                                  int stride, int extent) {
  int v = blockIdx.x * blockDim.x + threadIdx.x;
  if (v >= VOX) return;
  int pos = (v / stride) % extent;
  unsigned char ok = 1;
  if (pos < RAD || pos >= extent - RAD) {
    ok = 0;
  } else {
#pragma unroll
    for (int d = -RAD; d <= RAD; ++d) ok &= in[v + d * stride];
  }
  out[v] = ok;
}

// ---------------- fused hash + segment accumulation (TDM-staged) -------------
// LDS (dynamic): int sum0/1/2[4096]; uint cnt[4096]; float tile[3*CHUNK]
// = 64KB table + 48KB TDM tile = 114,688 B (well under 320KB/WGP).
__global__ void accumulate_kernel(const float* __restrict__ emb,
                                  const unsigned char* __restrict__ mask,
                                  int* __restrict__ gsum0, int* __restrict__ gsum1,
                                  int* __restrict__ gsum2, unsigned* __restrict__ gcnt) {
  extern __shared__ char smem[];
  int*      tsum0 = (int*)smem;
  int*      tsum1 = tsum0 + MCLUST;
  int*      tsum2 = tsum1 + MCLUST;
  unsigned* tcnt  = (unsigned*)(tsum2 + MCLUST);
  float*    tile  = (float*)(tcnt + MCLUST);

  for (int k = threadIdx.x; k < MCLUST; k += blockDim.x) {
    tsum0[k] = 0; tsum1[k] = 0; tsum2[k] = 0; tcnt[k] = 0u;
  }
  __syncthreads();

  for (int chunk = blockIdx.x; chunk < NCHUNK; chunk += gridDim.x) {
    const int base = chunk * CHUNK;

#ifdef HAVE_TDM
    // One wave per block drives the TDM: 2-D tile, dim0 = CHUNK voxels,
    // dim1 = 3 channels, row stride = VOX elements, 4B elements.
    if ((threadIdx.x >> 5) == 0) {
      unsigned ldsAddr = (unsigned)(uintptr_t)tile;       // low 32b == LDS offset
      uint64_t ga = (uint64_t)(uintptr_t)(emb + base);    // tile start in memory
      // D# group 0: count=1 | lds_addr | global_addr[56:0] | type=2
      u32x4 g0 = { 1u,
                   ldsAddr,
                   (unsigned)(ga & 0xFFFFFFFFu),
                   (unsigned)((ga >> 32) & 0x1FFFFFFu) | (2u << 30) };
      // D# group 1: data_size=4B; tensor_dim0=CHUNK; tensor_dim1=3;
      // tile_dim0=CHUNK; tile_dim1=3; tensor_dim0_stride=VOX.
      i32x8 g1 = { (int)(2u << 16),
                   (int)((CHUNK & 0xFFFFu) << 16),
                   (int)((CHUNK >> 16) | (3u << 16)),
                   (int)((3u >> 16) | ((unsigned)CHUNK << 16)),
                   (int)3,
                   (int)(VOX & 0xFFFFFFFFu),
                   (int)(((uint64_t)VOX >> 32) & 0xFFFFu),
                   0 };
      i32x4 gz = { 0, 0, 0, 0 };
#if defined(__clang_major__) && (__clang_major__ >= 23)
      i32x8 gz8 = { 0, 0, 0, 0, 0, 0, 0, 0 };
      __builtin_amdgcn_tensor_load_to_lds(g0, g1, gz, gz, gz8, 0);
#else
      __builtin_amdgcn_tensor_load_to_lds(g0, g1, gz, gz, 0);
#endif
      __builtin_amdgcn_s_wait_tensorcnt(0);
    }
#else
    for (int i = threadIdx.x; i < CHUNK; i += blockDim.x) {
      tile[i]             = emb[base + i];
      tile[CHUNK + i]     = emb[VOX + base + i];
      tile[2 * CHUNK + i] = emb[2 * VOX + base + i];
    }
#endif
    __syncthreads();

    for (int i = threadIdx.x; i < CHUNK; i += blockDim.x) {
      int v = base + i;
      unsigned char mk = mask[v];
      float e0 = tile[i], e1 = tile[CHUNK + i], e2 = tile[2 * CHUNK + i];
      // masked-out voxels become -10 => never valid; last voxel forced invalid
      bool valid = mk && (e0 > -2.0f || e1 > -2.0f || e2 > -2.0f) && (v != VOX - 1);
      if (valid) {
        float c0 = rintf(e0 * SCALE_F);
        float c1 = rintf(e1 * SCALE_F);
        float c2 = rintf(e2 * SCALE_F);
        unsigned q0 = (unsigned)fminf(fmaxf(c0 + QOFF, 0.0f), 255.0f);
        unsigned q1 = (unsigned)fminf(fmaxf(c1 + QOFF, 0.0f), 255.0f);
        unsigned q2 = (unsigned)fminf(fmaxf(c2 + QOFF, 0.0f), 255.0f);
        unsigned h = q0 * 73856093u ^ q1 * 19349663u ^ q2 * 83492791u;
        unsigned seg = h & (MCLUST - 1);   // 4096 is a power of two
        atomicAdd(&tcnt[seg], 1u);         // ds_add_u32: integer => deterministic
        atomicAdd(&tsum0[seg], (int)c0);
        atomicAdd(&tsum1[seg], (int)c1);
        atomicAdd(&tsum2[seg], (int)c2);
      }
    }
    __syncthreads();  // protect tile against next chunk's TDM (WAR)
  }

  // flush per-WGP table with integer global atomics (order-independent => exact)
  for (int k = threadIdx.x; k < MCLUST; k += blockDim.x) {
    unsigned c = tcnt[k];
    if (c) {
      atomicAdd(&gcnt[k], c);
      atomicAdd(&gsum0[k], tsum0[k]);
      atomicAdd(&gsum1[k], tsum1[k]);
      atomicAdd(&gsum2[k], tsum2[k]);
    }
  }
}

// ---------------- centroids, boxes, scores ----------------
__global__ void finalize_kernel(const int* __restrict__ gsum0, const int* __restrict__ gsum1,
                                const int* __restrict__ gsum2, const unsigned* __restrict__ gcnt,
                                float* __restrict__ out, float* __restrict__ scores,
                                float* __restrict__ b0, float* __restrict__ b1,
                                float* __restrict__ b2, float* __restrict__ b3) {
  int k = blockIdx.x * blockDim.x + threadIdx.x;
  if (k >= MCLUST) return;
  float fc  = (float)gcnt[k];
  float den = fmaxf(fc, 1.0f);
  float c0 = (float)gsum0[k] / den;
  float c1 = (float)gsum1[k] / den;
  float c2 = (float)gsum2[k] / den;
  out[k * 3 + 0] = c0;
  out[k * 3 + 1] = c1;
  out[k * 3 + 2] = c2;
  scores[k] = (fc >= MINS) ? fc : -1.0f;
  b0[k] = c0 - BOXH; b1[k] = c1 - BOXH; b2[k] = c0 + BOXH; b3[k] = c1 + BOXH;
}

// ---------------- stable bitonic sort + greedy NMS (one WGP) ----------------
// dyn smem: score[4096] + idx[4096] + 4x box[4096] + keep[4096] = 114,688 B
__global__ void sort_nms_kernel(const float* __restrict__ scores,
                                const float* __restrict__ b0g, const float* __restrict__ b1g,
                                const float* __restrict__ b2g, const float* __restrict__ b3g,
                                float* __restrict__ out) {
  extern __shared__ char smem[];
  float* ss = (float*)smem;
  int*   si = (int*)(ss + MCLUST);
  float* b0 = (float*)(si + MCLUST);
  float* b1 = b0 + MCLUST;
  float* b2 = b1 + MCLUST;
  float* b3 = b2 + MCLUST;
  int*   kp = (int*)(b3 + MCLUST);

  for (int i = threadIdx.x; i < MCLUST; i += blockDim.x) { ss[i] = scores[i]; si[i] = i; }
  __syncthreads();

  // Bitonic sort: descending by score, ties broken by ascending original index
  // (mirrors stable jnp.argsort(-scores)).
  for (int k = 2; k <= MCLUST; k <<= 1) {
    for (int j = k >> 1; j > 0; j >>= 1) {
      for (int i = threadIdx.x; i < MCLUST; i += blockDim.x) {
        int l = i ^ j;
        if (l > i) {
          bool desc = ((i & k) == 0);
          float sa = ss[i], sb = ss[l];
          int   ia = si[i], ib = si[l];
          bool good = (sa > sb) || (sa == sb && ia < ib);  // i-before-l, descending
          if (desc != good) { ss[i] = sb; ss[l] = sa; si[i] = ib; si[l] = ia; }
        }
      }
      __syncthreads();
    }
  }

  for (int i = threadIdx.x; i < MCLUST; i += blockDim.x) {
    int o = si[i];
    b0[i] = b0g[o]; b1[i] = b1g[o]; b2[i] = b2g[o]; b3[i] = b3g[o];
    kp[i] = (ss[i] >= MINS) ? 1 : 0;   // is_cluster in sorted order
  }
  __syncthreads();

  for (int i = 0; i < MCLUST; ++i) {
    float x1i = b0[i], y1i = b1[i], x2i = b2[i], y2i = b3[i];
    int ki = kp[i];
    if (ki) {
      float a1 = (x2i - x1i) * (y2i - y1i);
      for (int t = threadIdx.x; t < MCLUST; t += blockDim.x) {
        if (t > i) {
          float x1 = fmaxf(x1i, b0[t]);
          float y1 = fmaxf(y1i, b1[t]);
          float x2 = fminf(x2i, b2[t]);
          float y2 = fminf(y2i, b3[t]);
          float inter = fmaxf(x2 - x1, 0.0f) * fmaxf(y2 - y1, 0.0f);
          float a2 = (b2[t] - b0[t]) * (b3[t] - b1[t]);
          float iou = inter / fmaxf(a1 + a2 - inter, 1e-9f);
          if (iou > IOUTHR) kp[t] = 0;
        }
      }
    }
    __syncthreads();
  }

  // scatter keep back to original cluster order; keep written as 0.0/1.0
  for (int i = threadIdx.x; i < MCLUST; i += blockDim.x) {
    out[MCLUST * 3 + si[i]] = kp[i] ? 1.0f : 0.0f;
  }
}

// ---------------- host glue ----------------
extern "C" void kernel_launch(void* const* d_in, const int* in_sizes, int n_in,
                              void* d_out, int out_size, void* d_ws, size_t ws_size,
                              hipStream_t stream) {
  const float* emb = (const float*)d_in[0];   // (3, 512*512*48)
  const float* pm  = (const float*)d_in[1];   // (512*512*48)
  float* out = (float*)d_out;                 // 4096*3 centroids + 4096 keep

  char* ws = (char*)d_ws;
  unsigned char* A = (unsigned char*)ws;               // VOX bytes
  unsigned char* B = (unsigned char*)(ws + VOX);       // VOX bytes
  char* tbl = ws + 2 * (size_t)VOX;
  int*      gsum0 = (int*)tbl;
  int*      gsum1 = gsum0 + MCLUST;
  int*      gsum2 = gsum1 + MCLUST;
  unsigned* gcnt  = (unsigned*)(gsum2 + MCLUST);
  float* scores = (float*)(tbl + 65536);
  float* b0 = scores + MCLUST;
  float* b1 = b0 + MCLUST;
  float* b2 = b1 + MCLUST;
  float* b3 = b2 + MCLUST;

  hipMemsetAsync(tbl, 0, 65536, stream);  // zero the integer accumulation table

  const int T = 256;
  const int G = VOX / T;
  // fused binarize + w-axis erosion on the WMMA path (saves one 12.6MB pass)
  erode_w_wmma_kernel<<<1024, T, 0, stream>>>(pm, A);
  erode_axis_kernel<<<G, T, 0, stream>>>(A, B, D2, D1);         // along x (512)
  erode_axis_kernel<<<G, T, 0, stream>>>(B, A, D2 * D1, D0);    // along y (512)

  const size_t accumSmem = (size_t)MCLUST * 4 * 4 + (size_t)3 * CHUNK * 4; // 114,688
  accumulate_kernel<<<512, 256, accumSmem, stream>>>(emb, A, gsum0, gsum1, gsum2, gcnt);

  finalize_kernel<<<MCLUST / 256, 256, 0, stream>>>(gsum0, gsum1, gsum2, gcnt,
                                                    out, scores, b0, b1, b2, b3);

  const size_t nmsSmem = (size_t)MCLUST * 7 * 4;                           // 114,688
  sort_nms_kernel<<<1, 1024, nmsSmem, stream>>>(scores, b0, b1, b2, b3, out);
}
